// ChamferLoss_45329084842106
// MI455X (gfx1250) — matrix operand order repaired and checked by reading, not compiled, not run on hardware
//
#include <hip/hip_runtime.h>

#define B_   32
#define D_   128
#define N_   2048
#define TM   128          // x rows per workgroup
#define TN   128          // y cols per tile step
#define LDSS 136          // padded LDS row stride (bf16 elems): 272B = 17*16B, 8B-divisible
#define BIGF 1e30f

typedef __attribute__((ext_vector_type(16))) __bf16 v16bf;
typedef __attribute__((ext_vector_type(8)))  __bf16 v8bf;
typedef __attribute__((ext_vector_type(4)))  __bf16 v4bf;
typedef __attribute__((ext_vector_type(8)))  float  v8f;

// Load a 16-element bf16 operand as two 16B chunks (both 16B aligned in LDS).
__device__ __forceinline__ v16bf load_op(const __bf16* base, int second_off) {
  union { struct { v8bf a, b; } p; v16bf v; } u;
  u.p.a = *(const v8bf*)(base);
  u.p.b = *(const v8bf*)(base + second_off);
  return u.v;
}

// Convert a 128x128 fp32 tile (source layout [D][N], column block col0) into
// split-bf16 hi/lo LDS tiles laid out [point][D] (row stride LDSS).
// 256 threads: point p = tid&127 (lane-consecutive -> coalesced global reads),
// d-half = tid>>7. Each thread stores contiguous packed b64 rows.
__device__ __forceinline__ void load_tile(const float* __restrict__ src, int col0,
                                          __bf16* __restrict__ sh,
                                          __bf16* __restrict__ sl, int tid) {
  const int p  = tid & 127;
  const int d0 = (tid >> 7) * 64;
  const float* g = src + (size_t)d0 * N_ + col0 + p;
  __bf16* ph = sh + p * LDSS + d0;
  __bf16* pl = sl + p * LDSS + d0;
#pragma unroll
  for (int dd = 0; dd < 64; dd += 4) {
    float f0 = g[(size_t)(dd + 0) * N_];
    float f1 = g[(size_t)(dd + 1) * N_];
    float f2 = g[(size_t)(dd + 2) * N_];
    float f3 = g[(size_t)(dd + 3) * N_];
    __bf16 h0 = (__bf16)f0, h1 = (__bf16)f1, h2 = (__bf16)f2, h3 = (__bf16)f3;
    __bf16 l0 = (__bf16)(f0 - (float)h0);
    __bf16 l1 = (__bf16)(f1 - (float)h1);
    __bf16 l2 = (__bf16)(f2 - (float)h2);
    __bf16 l3 = (__bf16)(f3 - (float)h3);
    v4bf hv = {h0, h1, h2, h3};
    v4bf lv = {l0, l1, l2, l3};
    *(v4bf*)(ph + dd) = hv;   // 8B-aligned: p*272 + 2*(d0+dd)
    *(v4bf*)(pl + dd) = lv;
  }
}

// ---------------- kernel 1: norms + init ----------------
__global__ void chamfer_prep(const float* __restrict__ x, const float* __restrict__ y,
                             float* __restrict__ x2, float* __restrict__ y2,
                             unsigned* __restrict__ rowmin, unsigned* __restrict__ colmin,
                             float* __restrict__ out) {
  int idx = blockIdx.x * blockDim.x + threadIdx.x;   // 0 .. B*N-1 exactly
  if (idx == 0) out[0] = 0.0f;
  int b = idx / N_, n = idx % N_;
  const float* xp = x + (size_t)b * D_ * N_ + n;
  const float* yp = y + (size_t)b * D_ * N_ + n;
  float sx = 0.f, sy = 0.f;
#pragma unroll 8
  for (int d = 0; d < D_; ++d) {
    float xv = xp[(size_t)d * N_]; sx = fmaf(xv, xv, sx);
    float yv = yp[(size_t)d * N_]; sy = fmaf(yv, yv, sy);
  }
  x2[idx] = sx; y2[idx] = sy;
  unsigned big = __float_as_uint(BIGF);
  rowmin[idx] = big; colmin[idx] = big;
}

// ---------------- kernel 2: WMMA GEMM + fused chamfer mins ----------------
__global__ __launch_bounds__(256)
void chamfer_gemm(const float* __restrict__ x, const float* __restrict__ y,
                  const unsigned char* __restrict__ mask,
                  const float* __restrict__ x2g, const float* __restrict__ y2g,
                  unsigned* __restrict__ rowmin, unsigned* __restrict__ colmin) {
  __shared__ __bf16 sXh[TM * LDSS];
  __shared__ __bf16 sXl[TM * LDSS];
  __shared__ __bf16 sYh[TN * LDSS];
  __shared__ __bf16 sYl[TN * LDSS];

  const int b    = blockIdx.x / (N_ / TM);
  const int i0   = (blockIdx.x % (N_ / TM)) * TM;
  const int tid  = threadIdx.x;
  const int lane = tid & 31;
  const int w    = tid >> 5;
  const int kh   = lane >> 4;      // lane half
  const int l15  = lane & 15;
  // 8 waves in a 4x2 grid over the 8x8 (16x16)-tile output block:
  const int mrow = w >> 1;         // 0..3 -> 32-row group (two 16-row strips)
  const int tgrp = w & 1;          // 0..1 -> four 16-col tiles each

  const float* xb = x + (size_t)b * D_ * N_;
  const float* yb = y + (size_t)b * D_ * N_;

  // ---- X tile (loaded once) ----
  load_tile(xb, i0, sXh, sXl, tid);

  // per-lane row metadata: rows i = i0 + 32*mrow + 16*s + 8*kh + r
  float x2v[2][8], xpen[2][8], rmin[2][8];
#pragma unroll
  for (int s = 0; s < 2; ++s)
#pragma unroll
    for (int r = 0; r < 8; ++r) {
      int gi = i0 + 32 * mrow + 16 * s + 8 * kh + r;
      x2v[s][r]  = x2g[b * N_ + gi];
      xpen[s][r] = mask[b * N_ + gi] ? 0.0f : BIGF;
      rmin[s][r] = BIGF;
    }

  const v8f zacc = {0.f, 0.f, 0.f, 0.f, 0.f, 0.f, 0.f, 0.f};

  for (int jt = 0; jt < N_ / TN; ++jt) {
    const int j0 = jt * TN;
    __syncthreads();               // previous-iteration readers done
    load_tile(yb, j0, sYh, sYl, tid);
    __syncthreads();

    // per-lane column metadata (this wave's cols: j = j0 + 16*(4*tgrp+t) + l15)
    float y2v[4], ypen[4];
#pragma unroll
    for (int t = 0; t < 4; ++t) {
      int gj  = j0 + 16 * (4 * tgrp + t) + l15;
      y2v[t]  = y2g[b * N_ + gj];
      ypen[t] = mask[b * N_ + gj] ? 0.0f : BIGF;
    }

    v8f acc0[4], acc1[4];
#pragma unroll
    for (int t = 0; t < 4; ++t) { acc0[t] = zacc; acc1[t] = zacc; }

    // ---- K loop: split-bf16 3-term GEMM, two interleaved row strips ----
#pragma unroll
    for (int k0 = 0; k0 < D_; k0 += 32) {
      const int a0 = (32 * mrow + l15) * LDSS + k0 + 8 * kh;
      const int a1 = a0 + 16 * LDSS;
      v16bf Ah0 = load_op(&sXh[a0], 16);
      v16bf Al0 = load_op(&sXl[a0], 16);
      v16bf Ah1 = load_op(&sXh[a1], 16);
      v16bf Al1 = load_op(&sXl[a1], 16);
#pragma unroll
      for (int t = 0; t < 4; ++t) {
        const int bo = (16 * (4 * tgrp + t) + l15) * LDSS + k0 + 8 * kh;
        v16bf Bh = load_op(&sYh[bo], 16);
        v16bf Bl = load_op(&sYl[bo], 16);
        // interleave the two strips -> dependency distance 2 between WMMAs
        acc0[t] = __builtin_amdgcn_wmma_f32_16x16x32_bf16(false, Ah0, false, Bh,
                                                          (short)0, acc0[t], false, false);
        acc1[t] = __builtin_amdgcn_wmma_f32_16x16x32_bf16(false, Ah1, false, Bh,
                                                          (short)0, acc1[t], false, false);
        acc0[t] = __builtin_amdgcn_wmma_f32_16x16x32_bf16(false, Ah0, false, Bl,
                                                          (short)0, acc0[t], false, false);
        acc1[t] = __builtin_amdgcn_wmma_f32_16x16x32_bf16(false, Ah1, false, Bl,
                                                          (short)0, acc1[t], false, false);
        acc0[t] = __builtin_amdgcn_wmma_f32_16x16x32_bf16(false, Al0, false, Bh,
                                                          (short)0, acc0[t], false, false);
        acc1[t] = __builtin_amdgcn_wmma_f32_16x16x32_bf16(false, Al1, false, Bh,
                                                          (short)0, acc1[t], false, false);
      }
    }

    // ---- fused epilogue: d = x2 + y2 - 2*dot (+BIG penalties for mask) ----
#pragma unroll
    for (int t = 0; t < 4; ++t) {
      float cm = BIGF;
#pragma unroll
      for (int s = 0; s < 2; ++s) {
        const v8f a = s ? acc1[t] : acc0[t];
#pragma unroll
        for (int r = 0; r < 8; ++r) {
          float dg = fmaxf(x2v[s][r] + y2v[t] - 2.0f * a[r], 0.0f);
          rmin[s][r] = fminf(rmin[s][r], dg + ypen[t]);  // min over valid j
          cm         = fminf(cm,         dg + xpen[s][r]); // min over valid i
        }
      }
      cm = fminf(cm, __shfl_xor(cm, 16, 32));            // combine row halves
      if (kh == 0)
        atomicMin(&colmin[b * N_ + j0 + 16 * (4 * tgrp + t) + l15],
                  __float_as_uint(cm));
    }
  }

  // ---- final row-min: cross-lane reduce within 16-lane halves ----
#pragma unroll
  for (int s = 0; s < 2; ++s)
#pragma unroll
    for (int r = 0; r < 8; ++r) {
      float v = rmin[s][r];
      v = fminf(v, __shfl_xor(v, 1, 32));
      v = fminf(v, __shfl_xor(v, 2, 32));
      v = fminf(v, __shfl_xor(v, 4, 32));
      v = fminf(v, __shfl_xor(v, 8, 32));
      if (l15 == 0)
        atomicMin(&rowmin[b * N_ + i0 + 32 * mrow + 16 * s + 8 * kh + r],
                  __float_as_uint(v));
    }
}

// ---------------- kernel 3: masked sums + mean ----------------
__global__ void chamfer_finish(const unsigned char* __restrict__ mask,
                               const unsigned* __restrict__ rowmin,
                               const unsigned* __restrict__ colmin,
                               float* __restrict__ out) {
  __shared__ float red[256];
  int b = blockIdx.x;
  float s = 0.f;
  for (int n = threadIdx.x; n < N_; n += 256) {
    int idx = b * N_ + n;
    if (mask[idx])
      s += __uint_as_float(rowmin[idx]) + __uint_as_float(colmin[idx]);
  }
  red[threadIdx.x] = s;
  __syncthreads();
  for (int st = 128; st > 0; st >>= 1) {
    if (threadIdx.x < st) red[threadIdx.x] += red[threadIdx.x + st];
    __syncthreads();
  }
  if (threadIdx.x == 0) atomicAdd(out, red[0] * (1.0f / (float)B_));
}

extern "C" void kernel_launch(void* const* d_in, const int* in_sizes, int n_in,
                              void* d_out, int out_size, void* d_ws, size_t ws_size,
                              hipStream_t stream) {
  (void)in_sizes; (void)n_in; (void)out_size; (void)ws_size;
  const float*         x    = (const float*)d_in[0];
  const float*         y    = (const float*)d_in[1];
  const unsigned char* mask = (const unsigned char*)d_in[2];  // jnp bool = 1 byte
  float* out = (float*)d_out;

  const int BN = B_ * N_;
  float*    x2     = (float*)d_ws;
  float*    y2     = x2 + BN;
  unsigned* rowmin = (unsigned*)(y2 + BN);
  unsigned* colmin = rowmin + BN;

  chamfer_prep<<<BN / 256, 256, 0, stream>>>(x, y, x2, y2, rowmin, colmin, out);
  chamfer_gemm<<<B_ * (N_ / TM), 256, 0, stream>>>(x, y, mask, x2, y2, rowmin, colmin);
  chamfer_finish<<<B_, 256, 0, stream>>>(mask, rowmin, colmin, out);
}
